// MoeConformerEncoder_7155415515313
// MI455X (gfx1250) — compile-verified, matrix operand order checked by reading, not run on hardware
//
#include <hip/hip_runtime.h>
#include <math.h>

// ---------------- model constants ----------------
#define B_    16
#define T_    2048
#define FIN_  80
#define D_    512
#define H_    8
#define DK_   64
#define FF_   2048
#define L_    6
#define E_    8
#define EH_   2048
#define KC_   15
#define TP_   511
#define NTOK_ (B_ * TP_)          // 8176
#define CAP_  1277                // int(1.25*NTOK/E)
#define XSCALE_ 22.627416997969522f

// ---------------- types ----------------
typedef __bf16 v16bf __attribute__((ext_vector_type(16)));
typedef float  v8f   __attribute__((ext_vector_type(8)));
typedef unsigned int uint4v __attribute__((ext_vector_type(4)));

union FragU { v16bf bf; uint4v u4[2]; };

static __device__ __forceinline__ unsigned short f2bf(float f) {
  union { float f; unsigned int u; } c; c.f = f;
  unsigned int u = c.u;
  u = u + 0x7FFFu + ((u >> 16) & 1u);   // round-to-nearest-even
  return (unsigned short)(u >> 16);
}
static __device__ __forceinline__ float swishf(float v) {
  return v / (1.f + expf(-v));
}

// =====================================================================
// Fast batched GEMM (both operands K-contiguous, rows 16B aligned):
//   C = act((A*B + bias)*alpha)*oscale [+ Cold]
// A: MxK bf16, element (m,k) at A[m*sAm + k]
// B: KxN bf16, element (k,n) at B[n*sBn + k]   (i.e. B stored N-major)
// C: row stride ldc, col stride 1.  grid.z batches (strides bA/bB/bC/bBias).
// Workgroup = 128x64 tile, 8 waves, each wave 32x32 -> 4 x
// v_wmma_f32_16x16x32_bf16 per K-step of 32.  Double-buffered LDS with
// b128 global loads / b128 LDS fills; one barrier per K-step.
// Requirements: sAm,sBn multiples of 8; K multiple of 32 OR the B rows are
// zero-padded over the K tail (used by the attn*V GEMM, K=511).
// Row indices are clamped instead of predicated: OOB rows only feed outputs
// that are never stored.
// =====================================================================
__global__ __launch_bounds__(256) void k_gemm(
    const unsigned short* __restrict__ A, long long sAm, long long bA,
    const unsigned short* __restrict__ B, long long sBn, long long bB,
    float* __restrict__ Cf, unsigned short* __restrict__ Cbf, long long ldc, long long bC,
    const float* __restrict__ bias, long long bBias,
    int M, int N, int K, int accum, int act, float alpha, float oscale)
{
  __shared__ __align__(16) unsigned short As[2][128 * 40];  // 40-half row pad: 80B = 5*16
  __shared__ __align__(16) unsigned short Bs[2][64 * 40];

  const int z = blockIdx.z;
  A += (long long)z * bA;
  B += (long long)z * bB;
  const long long coff = (long long)z * bC;
  const float* bp = bias ? (bias + (long long)z * bBias) : (const float*)0;

  const int m0 = blockIdx.y * 128;
  const int n0 = blockIdx.x * 64;
  const int tid  = threadIdx.x;
  const int lane = tid & 31;
  const int w    = tid >> 5;
  const int wrow = w >> 1;      // 0..3 -> 32-row strip
  const int wcol = w & 1;       // 0..1 -> 32-col strip
  const int lh   = lane >> 4;
  const int lr   = lane & 15;

  // tile-load addressing: each thread owns one b128 octet in two A rows + one B row
  const int oct = tid & 3;      // which 8-half group in the 32-wide K slice
  const int r0  = tid >> 2;     // 0..63
  int am0 = m0 + r0;        if (am0 > M - 1) am0 = M - 1;
  int am1 = m0 + 64 + r0;   if (am1 > M - 1) am1 = M - 1;
  int bn  = n0 + r0;        if (bn  > N - 1) bn  = N - 1;
  const unsigned short* Ap0 = A + (long long)am0 * sAm + oct * 8;
  const unsigned short* Ap1 = A + (long long)am1 * sAm + oct * 8;
  const unsigned short* Bp  = B + (long long)bn  * sBn + oct * 8;
  const int lo = r0 * 40 + oct * 8;

  v8f acc00 = {}, acc01 = {}, acc10 = {}, acc11 = {};

  // preload K-step 0
  uint4v va0 = *(const uint4v*)Ap0;
  uint4v va1 = *(const uint4v*)Ap1;
  uint4v vb  = *(const uint4v*)Bp;
  *(uint4v*)&As[0][lo]           = va0;
  *(uint4v*)&As[0][64 * 40 + lo] = va1;
  *(uint4v*)&Bs[0][lo]           = vb;
  __syncthreads();

  int stage = 0;
  for (int k0 = 0; k0 < K; k0 += 32) {
    const int kn = k0 + 32;
    const bool nxt = kn < K;          // uniform -> scalar branch
    if (nxt) {
      va0 = *(const uint4v*)(Ap0 + kn);
      va1 = *(const uint4v*)(Ap1 + kn);
      vb  = *(const uint4v*)(Bp  + kn);
    }
    FragU fa0, fa1, fb0, fb1;
    {
      const unsigned short* a0 = &As[stage][(wrow * 32 + lr) * 40];
      const unsigned short* a1 = &As[stage][(wrow * 32 + 16 + lr) * 40];
      const unsigned short* b0 = &Bs[stage][(wcol * 32 + lr) * 40];
      const unsigned short* b1 = &Bs[stage][(wcol * 32 + 16 + lr) * 40];
      const int o0 = 8 * lh, o1 = 16 + 8 * lh;
      fa0.u4[0] = *(const uint4v*)(a0 + o0); fa0.u4[1] = *(const uint4v*)(a0 + o1);
      fa1.u4[0] = *(const uint4v*)(a1 + o0); fa1.u4[1] = *(const uint4v*)(a1 + o1);
      fb0.u4[0] = *(const uint4v*)(b0 + o0); fb0.u4[1] = *(const uint4v*)(b0 + o1);
      fb1.u4[0] = *(const uint4v*)(b1 + o0); fb1.u4[1] = *(const uint4v*)(b1 + o1);
    }
    acc00 = __builtin_amdgcn_wmma_f32_16x16x32_bf16(false, fa0.bf, false, fb0.bf,
                                                    (short)0, acc00, false, false);
    acc01 = __builtin_amdgcn_wmma_f32_16x16x32_bf16(false, fa0.bf, false, fb1.bf,
                                                    (short)0, acc01, false, false);
    acc10 = __builtin_amdgcn_wmma_f32_16x16x32_bf16(false, fa1.bf, false, fb0.bf,
                                                    (short)0, acc10, false, false);
    acc11 = __builtin_amdgcn_wmma_f32_16x16x32_bf16(false, fa1.bf, false, fb1.bf,
                                                    (short)0, acc11, false, false);
    if (nxt) {
      const int ns = stage ^ 1;
      *(uint4v*)&As[ns][lo]           = va0;
      *(uint4v*)&As[ns][64 * 40 + lo] = va1;
      *(uint4v*)&Bs[ns][lo]           = vb;
    }
    __syncthreads();
    stage ^= 1;
  }

  // epilogue: C layout per 16x16 tile: VGPR i -> M = i + 8*lh, N = lr
  auto epi = [&](v8f& acc, int mbase, int nbase) {
#pragma unroll
    for (int i = 0; i < 8; ++i) {
      int m = mbase + i + 8 * lh;
      int n = nbase + lr;
      if (m < M && n < N) {
        float v = acc[i];
        if (bp) v += bp[n];
        v *= alpha;
        if (act == 1) v = fmaxf(v, 0.f);
        else if (act == 2) v = swishf(v);
        v *= oscale;
        long long off = coff + (long long)m * ldc + n;
        if (accum && Cf) v += Cf[off];
        if (Cf)  Cf[off]  = v;
        if (Cbf) Cbf[off] = f2bf(v);
      }
    }
  };
  epi(acc00, m0 + wrow * 32,      n0 + wcol * 32);
  epi(acc01, m0 + wrow * 32,      n0 + wcol * 32 + 16);
  epi(acc10, m0 + wrow * 32 + 16, n0 + wcol * 32);
  epi(acc11, m0 + wrow * 32 + 16, n0 + wcol * 32 + 16);
}

// ---------------- cast / transpose kernels ----------------

__global__ void k_cast(const float* __restrict__ s, unsigned short* __restrict__ d, long long n) {
  long long i = (long long)blockIdx.x * 256 + threadIdx.x;
  if (i < n) d[i] = f2bf(s[i]);
}

// batched transpose-cast: w[e][k][n] (f32) -> wt[e][n][k] (bf16)
__global__ void k_castT(const float* __restrict__ wsrc, unsigned short* __restrict__ wt,
                        int K, int N, int nb) {
  long long total = (long long)nb * K * N;
  long long i = (long long)blockIdx.x * 256 + threadIdx.x;
  if (i >= total) return;
  long long per = (long long)K * N;
  int e = (int)(i / per);
  long long rem = i - (long long)e * per;
  int n = (int)(rem / K);
  int k = (int)(rem - (long long)n * K);
  wt[i] = f2bf(wsrc[(long long)e * per + (long long)k * N + n]);
}

// proj_w (c*19+f, n) -> pwT[n][f*512+c] bf16 so the proj GEMM sees a
// K-contiguous B matching the conv2 output stored (t, f, c).
__global__ void k_permw(const float* __restrict__ s, unsigned short* __restrict__ d) {
  long long i = (long long)blockIdx.x * 256 + threadIdx.x;
  if (i >= (long long)512 * 9728) return;
  int n = (int)(i / 9728);
  int kk = (int)(i - (long long)n * 9728);   // kk = f*512 + c
  int c = kk & 511;
  int f = kk >> 9;
  d[i] = f2bf(s[((size_t)c * 19 + f) * 512 + n]);
}

// v[b,s,h,d] bf16 -> vt[b][h][d][s] with row stride 512, pad col zeroed
// (the zero pad is the K=511 tail mask for the attn*V GEMM).
__global__ void k_vtrans(const unsigned short* __restrict__ v, unsigned short* __restrict__ vt) {
  long long idx = (long long)blockIdx.x * 256 + threadIdx.x;
  if (idx >= (long long)B_ * H_ * DK_ * 512) return;
  int s = (int)(idx & 511);
  int d = (int)((idx >> 9) & 63);
  int h = (int)((idx >> 15) & 7);
  int b = (int)(idx >> 18);
  vt[idx] = (s < TP_) ? v[((size_t)b * TP_ + s) * 512 + h * 64 + d] : (unsigned short)0;
}

// fused conv1(+bias,relu) -> im2col patch for conv2 implicit GEMM.
__global__ void k_im2col(const float* __restrict__ xsb, const float* __restrict__ w1,
                         const float* __restrict__ b1, unsigned short* __restrict__ out,
                         int t0, int nrows) {
  long long total = (long long)nrows * 19 * 4608;
  long long idx = (long long)blockIdx.x * 256 + threadIdx.x;
  if (idx >= total) return;
  int k = (int)(idx % 4608);
  long long m = idx / 4608;
  int f2 = (int)(m % 19);
  int t2 = t0 + (int)(m / 19);
  int ci = k / 9, r = k % 9, kh = r / 3, kw = r % 3;
  int h = 2 * t2 + kh, wcol = 2 * f2 + kw;
  const float* wp = w1 + (size_t)ci * 9;
  float acc = b1[ci];
#pragma unroll
  for (int ih = 0; ih < 3; ++ih)
#pragma unroll
    for (int iw = 0; iw < 3; ++iw)
      acc += xsb[(size_t)(2 * h + ih) * 80 + (2 * wcol + iw)] * wp[ih * 3 + iw];
  out[idx] = f2bf(fmaxf(acc, 0.f));
}

__global__ void k_mask_init(const int* __restrict__ lens, float* __restrict__ m01) {
  int idx = blockIdx.x * 256 + threadIdx.x;
  if (idx >= NTOK_) return;
  int b = idx / TP_, i = idx % TP_;
  m01[idx] = (4 * i < lens[b]) ? 1.f : 0.f;
}

__global__ void k_posenc(float* __restrict__ pef, unsigned short* __restrict__ pebf) {
  int idx = blockIdx.x * 256 + threadIdx.x;
  if (idx >= TP_ * 256) return;
  int t = idx >> 8, j = idx & 255;
  float arg = (float)t * expf(-(2.f * j) * logf(10000.f) / 512.f);
  float s = sinf(arg), c = cosf(arg);
  size_t o = (size_t)t * 512 + 2 * j;
  pef[o] = s; pef[o + 1] = c;
  pebf[o] = f2bf(s); pebf[o + 1] = f2bf(c);
}

// block-per-token LayerNorm; optional mask multiply, f32 and/or bf16 outputs.
__global__ __launch_bounds__(256) void k_ln(
    const float* __restrict__ x, const float* __restrict__ g, const float* __restrict__ b,
    const float* __restrict__ m01, float* __restrict__ yf, unsigned short* __restrict__ ybf)
{
  __shared__ float red[256];
  const long long t = blockIdx.x;
  const int tid = threadIdx.x;
  const float* row = x + t * 512;
  float a0 = row[tid], a1 = row[tid + 256];
  red[tid] = a0 + a1;
  __syncthreads();
  for (int s = 128; s > 0; s >>= 1) { if (tid < s) red[tid] += red[tid + s]; __syncthreads(); }
  float mean = red[0] * (1.f / 512.f);
  __syncthreads();
  float d0 = a0 - mean, d1 = a1 - mean;
  red[tid] = d0 * d0 + d1 * d1;
  __syncthreads();
  for (int s = 128; s > 0; s >>= 1) { if (tid < s) red[tid] += red[tid + s]; __syncthreads(); }
  float r = rsqrtf(red[0] * (1.f / 512.f) + 1e-5f);
  float mul = m01 ? m01[t] : 1.f;
  float y0 = (d0 * r * g[tid] + b[tid]) * mul;
  float y1 = (d1 * r * g[tid + 256] + b[tid + 256]) * mul;
  if (yf)  { yf[t * 512 + tid] = y0; yf[t * 512 + tid + 256] = y1; }
  if (ybf) { ybf[t * 512 + tid] = f2bf(y0); ybf[t * 512 + tid + 256] = f2bf(y1); }
}

__global__ void k_qubias(const float* __restrict__ qf, const float* __restrict__ bu,
                         const float* __restrict__ bv, unsigned short* __restrict__ qu,
                         unsigned short* __restrict__ qv) {
  long long idx = (long long)blockIdx.x * 256 + threadIdx.x;
  if (idx >= (long long)NTOK_ * 512) return;
  int c = (int)(idx & 511);
  float q = qf[idx];
  qu[idx] = f2bf(q + bu[c]);
  qv[idx] = f2bf(q + bv[c]);
}

// masked softmax over 511 keys (rows padded to 512); writes bf16 attn,
// zeroes the pad column (K-tail mask for the attn*V GEMM).
__global__ __launch_bounds__(256) void k_softmax(const float* __restrict__ sc,
                                                 unsigned short* __restrict__ attn,
                                                 const float* __restrict__ mb) {
  __shared__ float red[256];
  const long long row = blockIdx.x;           // h*511 + t
  const float* r = sc + row * 512;
  const int tid = threadIdx.x;
  const int c1 = tid + 256;
  float m0v = mb[tid];
  float v0 = (m0v > 0.f) ? r[tid] * 0.125f : -1e9f;
  float m1v = 0.f, v1 = -1e30f;
  if (c1 < TP_) { m1v = mb[c1]; v1 = (m1v > 0.f) ? r[c1] * 0.125f : -1e9f; }
  red[tid] = fmaxf(v0, v1);
  __syncthreads();
  for (int s = 128; s > 0; s >>= 1) { if (tid < s) red[tid] = fmaxf(red[tid], red[tid + s]); __syncthreads(); }
  float mx = red[0];
  __syncthreads();
  float e0 = expf(v0 - mx);
  float e1 = (c1 < TP_) ? expf(v1 - mx) : 0.f;
  red[tid] = e0 + e1;
  __syncthreads();
  for (int s = 128; s > 0; s >>= 1) { if (tid < s) red[tid] += red[tid + s]; __syncthreads(); }
  float inv = 1.f / red[0];
  unsigned short* o = attn + row * 512;
  o[tid] = f2bf((m0v > 0.f) ? e0 * inv : 0.f);
  if (c1 < TP_) o[c1] = f2bf((m1v > 0.f) ? e1 * inv : 0.f);
  if (tid == 0) o[TP_] = 0;   // zero pad column
}

__global__ void k_glu(const float* __restrict__ t2, float* __restrict__ out) {
  long long idx = (long long)blockIdx.x * 256 + threadIdx.x;
  if (idx >= (long long)NTOK_ * 512) return;
  long long tok = idx >> 9;
  int d = (int)(idx & 511);
  float a = t2[tok * 1024 + d];
  float g = t2[tok * 1024 + 512 + d];
  out[idx] = a * (1.f / (1.f + expf(-g)));
}

__global__ void k_dwconv(const float* __restrict__ in, const float* __restrict__ wd,
                         const float* __restrict__ db, const float* __restrict__ bg,
                         const float* __restrict__ bb, unsigned short* __restrict__ out) {
  long long idx = (long long)blockIdx.x * 256 + threadIdx.x;
  if (idx >= (long long)NTOK_ * 512) return;
  long long tok = idx >> 9;
  int d = (int)(idx & 511);
  int b = (int)(tok / TP_), t = (int)(tok % TP_);
  float acc = db[d];
#pragma unroll
  for (int k = 0; k < KC_; ++k) {
    int tt = t + k - 7;
    if (tt >= 0 && tt < TP_)
      acc += in[((size_t)b * TP_ + tt) * 512 + d] * wd[(size_t)d * KC_ + k];
  }
  acc = acc * bg[d] + bb[d];
  out[idx] = f2bf(swishf(acc));
}

__global__ void k_mask_add(float* __restrict__ x, const float* __restrict__ t,
                           const float* __restrict__ m01) {
  long long idx = (long long)blockIdx.x * 256 + threadIdx.x;
  if (idx >= (long long)NTOK_ * 512) return;
  x[idx] += t[idx] * m01[idx >> 9];
}

__global__ __launch_bounds__(256) void k_moe_gate(
    const float* __restrict__ flat, const float* __restrict__ gw,
    float* __restrict__ probs, int* __restrict__ eidx, float* __restrict__ pmax) {
  __shared__ float red[8 * 256];
  const long long t = blockIdx.x;
  const int tid = threadIdx.x;
  float part[8] = {0, 0, 0, 0, 0, 0, 0, 0};
  const float* row = flat + t * 512;
  for (int d = tid; d < 512; d += 256) {
    float xv = row[d];
    const float* g = gw + (size_t)d * 8;
#pragma unroll
    for (int e = 0; e < 8; ++e) part[e] += xv * g[e];
  }
#pragma unroll
  for (int e = 0; e < 8; ++e) red[e * 256 + tid] = part[e];
  __syncthreads();
  for (int s = 128; s > 0; s >>= 1) {
    if (tid < s)
#pragma unroll
      for (int e = 0; e < 8; ++e) red[e * 256 + tid] += red[e * 256 + tid + s];
    __syncthreads();
  }
  if (tid == 0) {
    float mx = -1e30f;
    for (int e = 0; e < 8; ++e) mx = fmaxf(mx, red[e * 256]);
    float p[8], sum = 0.f;
    for (int e = 0; e < 8; ++e) { p[e] = expf(red[e * 256] - mx); sum += p[e]; }
    int am = 0; float best = -1.f;
    for (int e = 0; e < 8; ++e) {
      float pr = p[e] / sum;
      probs[t * 8 + e] = pr;
      if (pr > best) { best = pr; am = e; }
    }
    eidx[t] = am;
    pmax[t] = best;
  }
}

__global__ void k_moe_scan(const int* __restrict__ eidx, int* __restrict__ posA) {
  int e = blockIdx.x;
  int c = 0;
  for (int i = 0; i < NTOK_; ++i)
    if (eidx[i] == e) { posA[i] = c; ++c; }
}

__global__ void k_moe_scatter(const float* __restrict__ flat, const int* __restrict__ eidx,
                              const int* __restrict__ posA, unsigned short* __restrict__ buf) {
  long long idx = (long long)blockIdx.x * 256 + threadIdx.x;
  if (idx >= (long long)NTOK_ * 512) return;
  long long tok = idx >> 9;
  int d = (int)(idx & 511);
  int p = posA[tok];
  if (p < CAP_) {
    int e = eidx[tok];
    buf[((size_t)e * CAP_ + p) * 512 + d] = f2bf(flat[idx]);
  }
}

__global__ void k_moe_combine(float* __restrict__ x, const float* __restrict__ flat,
                              const float* __restrict__ y, const int* __restrict__ eidx,
                              const int* __restrict__ posA, const float* __restrict__ pmax) {
  long long idx = (long long)blockIdx.x * 256 + threadIdx.x;
  if (idx >= (long long)NTOK_ * 512) return;
  long long tok = idx >> 9;
  int d = (int)(idx & 511);
  int p = posA[tok];
  int e = eidx[tok];
  float v = (p < CAP_) ? y[((size_t)e * CAP_ + p) * 512 + d] : flat[idx];
  x[idx] += 0.5f * pmax[tok] * v;
}

__global__ __launch_bounds__(256) void k_aux(const float* __restrict__ probs,
                                             const int* __restrict__ eidx,
                                             float* __restrict__ dst) {
  __shared__ float rp[8 * 256];
  __shared__ float rc[8 * 256];
  const int tid = threadIdx.x;
  float pp[8] = {0, 0, 0, 0, 0, 0, 0, 0};
  float cc[8] = {0, 0, 0, 0, 0, 0, 0, 0};
  for (int t = tid; t < NTOK_; t += 256) {
#pragma unroll
    for (int e = 0; e < 8; ++e) pp[e] += probs[(size_t)t * 8 + e];
    cc[eidx[t]] += 1.f;
  }
#pragma unroll
  for (int e = 0; e < 8; ++e) { rp[e * 256 + tid] = pp[e]; rc[e * 256 + tid] = cc[e]; }
  __syncthreads();
  for (int s = 128; s > 0; s >>= 1) {
    if (tid < s)
#pragma unroll
      for (int e = 0; e < 8; ++e) {
        rp[e * 256 + tid] += rp[e * 256 + tid + s];
        rc[e * 256 + tid] += rc[e * 256 + tid + s];
      }
    __syncthreads();
  }
  if (tid == 0) {
    float aux = 0.f;
    for (int e = 0; e < 8; ++e)
      aux += (rc[e * 256] / NTOK_) * (rp[e * 256] / NTOK_);
    *dst = 8.f * aux;
  }
}

__global__ void k_write_mask(float* __restrict__ dst, const float* __restrict__ m01) {
  int idx = blockIdx.x * 256 + threadIdx.x;
  if (idx < NTOK_) dst[idx] = m01[idx];
}

// ---------------- host side ----------------

static inline dim3 gsz(long long n) { return dim3((unsigned)((n + 255) / 256)); }

static void gemm(hipStream_t s,
                 const unsigned short* A, long long sAm, long long bA,
                 const unsigned short* B, long long sBn, long long bB,
                 float* Cf, unsigned short* Cbf, long long ldc, long long bC,
                 const float* bias, long long bBias,
                 int M, int N, int K, int nb, int accum, int act, float alpha, float oscale) {
  dim3 grid((unsigned)((N + 63) / 64), (unsigned)((M + 127) / 128), (unsigned)nb);
  k_gemm<<<grid, 256, 0, s>>>(A, sAm, bA, B, sBn, bB, Cf, Cbf, ldc, bC,
                              bias, bBias, M, N, K, accum, act, alpha, oscale);
}

extern "C" void kernel_launch(void* const* d_in, const int* in_sizes, int n_in,
                              void* d_out, int out_size, void* d_ws, size_t ws_size,
                              hipStream_t stream) {
  (void)in_sizes; (void)n_in; (void)out_size; (void)ws_size;

  const float* xs      = (const float*)d_in[0];
  const int*   lens    = (const int*)  d_in[1];
  const float* conv1_w = (const float*)d_in[2];
  const float* conv1_b = (const float*)d_in[3];
  const float* conv2_w = (const float*)d_in[4];
  const float* conv2_b = (const float*)d_in[5];
  const float* proj_w  = (const float*)d_in[6];
  const float* proj_b  = (const float*)d_in[7];
  const float* ln_g    = (const float*)d_in[8];
  const float* ln_b    = (const float*)d_in[9];
  const float* mac_w1  = (const float*)d_in[10];
  const float* mac_b1  = (const float*)d_in[11];
  const float* mac_w2  = (const float*)d_in[12];
  const float* mac_b2  = (const float*)d_in[13];
  const float* wq      = (const float*)d_in[14];
  const float* bq      = (const float*)d_in[15];
  const float* wk      = (const float*)d_in[16];
  const float* bk      = (const float*)d_in[17];
  const float* wv      = (const float*)d_in[18];
  const float* bv      = (const float*)d_in[19];
  const float* wo      = (const float*)d_in[20];
  const float* bo      = (const float*)d_in[21];
  const float* wpos    = (const float*)d_in[22];
  const float* bias_u  = (const float*)d_in[23];
  const float* bias_v  = (const float*)d_in[24];
  const float* pw1_w   = (const float*)d_in[25];
  const float* pw1_b   = (const float*)d_in[26];
  const float* dw_w    = (const float*)d_in[27];
  const float* dw_b    = (const float*)d_in[28];
  const float* bn_g    = (const float*)d_in[29];
  const float* bn_b    = (const float*)d_in[30];
  const float* pw2_w   = (const float*)d_in[31];
  const float* pw2_b   = (const float*)d_in[32];
  const float* gate_w  = (const float*)d_in[33];
  const float* e_w1    = (const float*)d_in[34];
  const float* e_b1    = (const float*)d_in[35];
  const float* e_w2    = (const float*)d_in[36];
  const float* e_b2    = (const float*)d_in[37];
  const float* oln_g   = (const float*)d_in[38];
  const float* oln_b   = (const float*)d_in[39];

  // ---- workspace bump allocator ----
  char* wp_ = (char*)d_ws;
  auto ALLOC = [&](size_t bytes) -> void* {
    void* r = (void*)wp_;
    wp_ += (bytes + 255) & ~(size_t)255;
    return r;
  };
  const size_t ND = (size_t)NTOK_ * 512;
  float*          x       = (float*)ALLOC(ND * 4);
  float*          lnf     = (float*)ALLOC(ND * 4);
  unsigned short* lnbf    = (unsigned short*)ALLOC(ND * 2);
  unsigned short* hbf     = (unsigned short*)ALLOC((size_t)NTOK_ * FF_ * 2);
  float*          tmp2D   = (float*)ALLOC((size_t)NTOK_ * 1024 * 4);
  float*          convx   = (float*)ALLOC(ND * 4);
  unsigned short* dwbf    = (unsigned short*)ALLOC(ND * 2);
  float*          tmpD    = (float*)ALLOC(ND * 4);
  float*          qf      = (float*)ALLOC(ND * 4);
  unsigned short* qu      = (unsigned short*)ALLOC(ND * 2);
  unsigned short* qv      = (unsigned short*)ALLOC(ND * 2);
  unsigned short* kbf     = (unsigned short*)ALLOC(ND * 2);
  unsigned short* vbf     = (unsigned short*)ALLOC(ND * 2);
  unsigned short* vtbf    = (unsigned short*)ALLOC((size_t)B_ * H_ * DK_ * 512 * 2);
  unsigned short* pkbf    = (unsigned short*)ALLOC((size_t)TP_ * 512 * 2);
  float*          pef     = (float*)ALLOC((size_t)TP_ * 512 * 4);
  unsigned short* pebf    = (unsigned short*)ALLOC((size_t)TP_ * 512 * 2);
  float*          sc_b    = (float*)ALLOC((size_t)H_ * TP_ * 512 * 4);   // rows padded to 512
  unsigned short* attn_b  = (unsigned short*)ALLOC((size_t)H_ * TP_ * 512 * 2);
  unsigned short* aobf    = (unsigned short*)ALLOC(ND * 2);
  unsigned short* wslot   = (unsigned short*)ALLOC((size_t)E_ * D_ * EH_ * 2);   // 16.8 MB
  unsigned short* im2c    = (unsigned short*)ALLOC((size_t)64 * 19 * 4608 * 2);
  unsigned short* c2out   = (unsigned short*)ALLOC((size_t)TP_ * 19 * 512 * 2);
  unsigned short* c2w     = (unsigned short*)ALLOC((size_t)512 * 4608 * 2);
  unsigned short* pwT     = (unsigned short*)ALLOC((size_t)512 * 9728 * 2);
  float*          m01     = (float*)ALLOC((size_t)NTOK_ * 4);
  float*          probs   = (float*)ALLOC((size_t)NTOK_ * 8 * 4);
  int*            eidx    = (int*)ALLOC((size_t)NTOK_ * 4);
  int*            posA    = (int*)ALLOC((size_t)NTOK_ * 4);
  float*          pmax    = (float*)ALLOC((size_t)NTOK_ * 4);
  unsigned short* buf     = (unsigned short*)ALLOC((size_t)E_ * CAP_ * 512 * 2);
  unsigned short* hmoe    = (unsigned short*)ALLOC((size_t)E_ * CAP_ * EH_ * 2);
  float*          ymoe    = (float*)ALLOC((size_t)E_ * CAP_ * 512 * 4);

  float* outf = (float*)d_out;
  float* aux_dst = outf + ND + NTOK_;

  // ---- mask + positional encoding ----
  k_mask_init<<<gsz(NTOK_), 256, 0, stream>>>(lens, m01);
  k_posenc<<<gsz((long long)TP_ * 256), 256, 0, stream>>>(pef, pebf);
  k_write_mask<<<gsz(NTOK_), 256, 0, stream>>>(outf + ND, m01);

  // ---- subsample: conv1 fused into im2col, conv2 implicit GEMM, proj GEMM ----
  // conv2_w is (co, ci*9) = (n, k) row-major -> already K-contiguous B.
  k_cast<<<gsz((long long)512 * 4608), 256, 0, stream>>>(conv2_w, c2w, (long long)512 * 4608);
  k_permw<<<gsz((long long)512 * 9728), 256, 0, stream>>>(proj_w, pwT);
  for (int b = 0; b < B_; ++b) {
    const float* xsb = xs + (size_t)b * T_ * FIN_;
    for (int ch = 0; ch < 8; ++ch) {
      int t0 = ch * 64;
      int nr = (TP_ - t0) < 64 ? (TP_ - t0) : 64;
      k_im2col<<<gsz((long long)nr * 19 * 4608), 256, 0, stream>>>(xsb, conv1_w, conv1_b, im2c, t0, nr);
      gemm(stream, im2c, 4608, 0, c2w, 4608, 0,
           nullptr, c2out + (size_t)t0 * 19 * 512, 512, 0,
           conv2_b, 0, nr * 19, 512, 4608, 1, 0, /*relu*/1, 1.f, 1.f);
    }
    gemm(stream, c2out, 9728, 0, pwT, 9728, 0,
         x + (size_t)b * TP_ * 512, nullptr, 512, 0,
         proj_b, 0, TP_, 512, 9728, 1, 0, 0, XSCALE_, 1.f);
  }

  // ---- layers ----
  for (int l = 0; l < L_; ++l) {
    const float* g0 = ln_g + ((size_t)l * 5 + 0) * 512; const float* c0 = ln_b + ((size_t)l * 5 + 0) * 512;
    const float* g1 = ln_g + ((size_t)l * 5 + 1) * 512; const float* c1 = ln_b + ((size_t)l * 5 + 1) * 512;
    const float* g2 = ln_g + ((size_t)l * 5 + 2) * 512; const float* c2 = ln_b + ((size_t)l * 5 + 2) * 512;
    const float* g3 = ln_g + ((size_t)l * 5 + 3) * 512; const float* c3 = ln_b + ((size_t)l * 5 + 3) * 512;
    const float* g4 = ln_g + ((size_t)l * 5 + 4) * 512; const float* c4 = ln_b + ((size_t)l * 5 + 4) * 512;

    // ---- macaron FFN: x += 0.5 * swish(ln(x)@w1+b1)@w2+b2 ----
    k_ln<<<NTOK_, 256, 0, stream>>>(x, g0, c0, nullptr, nullptr, lnbf);
    k_castT<<<gsz((long long)D_ * FF_), 256, 0, stream>>>(mac_w1 + (size_t)l * D_ * FF_, wslot, 512, FF_, 1);
    gemm(stream, lnbf, 512, 0, wslot, 512, 0, nullptr, hbf, FF_, 0,
         mac_b1 + (size_t)l * FF_, 0, NTOK_, FF_, 512, 1, 0, /*swish*/2, 1.f, 1.f);
    k_castT<<<gsz((long long)FF_ * D_), 256, 0, stream>>>(mac_w2 + (size_t)l * FF_ * D_, wslot, FF_, 512, 1);
    gemm(stream, hbf, FF_, 0, wslot, FF_, 0, x, nullptr, 512, 0,
         mac_b2 + (size_t)l * 512, 0, NTOK_, 512, FF_, 1, /*accum*/1, 0, 1.f, 0.5f);

    // ---- rel attention ----
    k_ln<<<NTOK_, 256, 0, stream>>>(x, g1, c1, nullptr, nullptr, lnbf);
    k_castT<<<gsz((long long)D_ * D_), 256, 0, stream>>>(wq + (size_t)l * D_ * D_, wslot, 512, 512, 1);
    gemm(stream, lnbf, 512, 0, wslot, 512, 0, qf, nullptr, 512, 0,
         bq + (size_t)l * 512, 0, NTOK_, 512, 512, 1, 0, 0, 1.f, 1.f);
    k_castT<<<gsz((long long)D_ * D_), 256, 0, stream>>>(wk + (size_t)l * D_ * D_, wslot, 512, 512, 1);
    gemm(stream, lnbf, 512, 0, wslot, 512, 0, nullptr, kbf, 512, 0,
         bk + (size_t)l * 512, 0, NTOK_, 512, 512, 1, 0, 0, 1.f, 1.f);
    k_castT<<<gsz((long long)D_ * D_), 256, 0, stream>>>(wv + (size_t)l * D_ * D_, wslot, 512, 512, 1);
    gemm(stream, lnbf, 512, 0, wslot, 512, 0, nullptr, vbf, 512, 0,
         bv + (size_t)l * 512, 0, NTOK_, 512, 512, 1, 0, 0, 1.f, 1.f);
    k_qubias<<<gsz((long long)ND), 256, 0, stream>>>(qf, bias_u + (size_t)l * 512,
                                                     bias_v + (size_t)l * 512, qu, qv);
    k_vtrans<<<gsz((long long)B_ * H_ * DK_ * 512), 256, 0, stream>>>(vbf, vtbf);
    k_castT<<<gsz((long long)D_ * D_), 256, 0, stream>>>(wpos + (size_t)l * D_ * D_, wslot, 512, 512, 1);
    gemm(stream, pebf, 512, 0, wslot, 512, 0, nullptr, pkbf, 512, 0,
         nullptr, 0, TP_, 512, 512, 1, 0, 0, 1.f, 1.f);

    for (int b = 0; b < B_; ++b) {
      const size_t ob = (size_t)b * TP_ * 512;
      // ac = (q+u) k^T  (head-batched over grid.z; K=64 contiguous in d)
      gemm(stream, qu + ob, 512, 64, kbf + ob, 512, 64,
           sc_b, nullptr, 512, (long long)TP_ * 512, nullptr, 0,
           TP_, TP_, DK_, H_, 0, 0, 1.f, 1.f);
      // bd = (q+v) pk^T  accumulated
      gemm(stream, qv + ob, 512, 64, pkbf, 512, 64,
           sc_b, nullptr, 512, (long long)TP_ * 512, nullptr, 0,
           TP_, TP_, DK_, H_, 1, 0, 1.f, 1.f);
      k_softmax<<<H_ * TP_, 256, 0, stream>>>(sc_b, attn_b, m01 + (size_t)b * TP_);
      // out = attn @ v  (K=511; pad column of attn/vt is zeroed)
      gemm(stream, attn_b, 512, (long long)TP_ * 512,
           vtbf + (size_t)b * H_ * DK_ * 512, 512, (long long)DK_ * 512,
           nullptr, aobf + ob, 512, 64, nullptr, 0,
           TP_, DK_, TP_, H_, 0, 0, 1.f, 1.f);
    }
    k_castT<<<gsz((long long)D_ * D_), 256, 0, stream>>>(wo + (size_t)l * D_ * D_, wslot, 512, 512, 1);
    gemm(stream, aobf, 512, 0, wslot, 512, 0, x, nullptr, 512, 0,
         bo + (size_t)l * 512, 0, NTOK_, 512, 512, 1, /*accum*/1, 0, 1.f, 1.f);

    // ---- conv module ----
    k_ln<<<NTOK_, 256, 0, stream>>>(x, g2, c2, m01, nullptr, lnbf);
    k_castT<<<gsz((long long)D_ * 1024), 256, 0, stream>>>(pw1_w + (size_t)l * D_ * 1024, wslot, 512, 1024, 1);
    gemm(stream, lnbf, 512, 0, wslot, 512, 0, tmp2D, nullptr, 1024, 0,
         pw1_b + (size_t)l * 1024, 0, NTOK_, 1024, 512, 1, 0, 0, 1.f, 1.f);
    k_glu<<<gsz((long long)ND), 256, 0, stream>>>(tmp2D, convx);
    k_dwconv<<<gsz((long long)ND), 256, 0, stream>>>(convx, dw_w + (size_t)l * D_ * KC_,
                                                     dw_b + (size_t)l * 512,
                                                     bn_g + (size_t)l * 512,
                                                     bn_b + (size_t)l * 512, dwbf);
    k_castT<<<gsz((long long)D_ * D_), 256, 0, stream>>>(pw2_w + (size_t)l * D_ * D_, wslot, 512, 512, 1);
    gemm(stream, dwbf, 512, 0, wslot, 512, 0, tmpD, nullptr, 512, 0,
         pw2_b + (size_t)l * 512, 0, NTOK_, 512, 512, 1, 0, 0, 1.f, 1.f);
    k_mask_add<<<gsz((long long)ND), 256, 0, stream>>>(x, tmpD, m01);

    // ---- MoE ----
    k_ln<<<NTOK_, 256, 0, stream>>>(x, g3, c3, nullptr, lnf, nullptr);
    k_moe_gate<<<NTOK_, 256, 0, stream>>>(lnf, gate_w + (size_t)l * D_ * E_, probs, eidx, pmax);
    k_moe_scan<<<E_, 1, 0, stream>>>(eidx, posA);
    k_moe_scatter<<<gsz((long long)ND), 256, 0, stream>>>(lnf, eidx, posA, buf);
    k_castT<<<gsz((long long)E_ * D_ * EH_), 256, 0, stream>>>(e_w1 + (size_t)l * E_ * D_ * EH_,
                                                               wslot, 512, EH_, E_);
    gemm(stream, buf, 512, (long long)CAP_ * 512,
         wslot, 512, (long long)EH_ * 512,
         nullptr, hmoe, EH_, (long long)CAP_ * EH_,
         e_b1 + (size_t)l * E_ * EH_, EH_,
         CAP_, EH_, 512, E_, 0, /*swish*/2, 1.f, 1.f);
    k_castT<<<gsz((long long)E_ * EH_ * D_), 256, 0, stream>>>(e_w2 + (size_t)l * E_ * EH_ * D_,
                                                               wslot, EH_, 512, E_);
    gemm(stream, hmoe, EH_, (long long)CAP_ * EH_,
         wslot, EH_, (long long)512 * EH_,
         ymoe, nullptr, 512, (long long)CAP_ * 512,
         e_b2 + (size_t)l * E_ * 512, 512,
         CAP_, 512, EH_, E_, 0, 0, 1.f, 1.f);
    k_moe_combine<<<gsz((long long)ND), 256, 0, stream>>>(x, lnf, ymoe, eidx, posA, pmax);
    k_aux<<<1, 256, 0, stream>>>(probs, eidx, aux_dst + l);

    // ---- end-of-layer LN (in place) ----
    k_ln<<<NTOK_, 256, 0, stream>>>(x, g4, c4, nullptr, x, nullptr);
  }

  // ---- final LN straight into d_out ----
  k_ln<<<NTOK_, 256, 0, stream>>>(x, oln_g, oln_b, nullptr, outf, nullptr);
}
